// Attention_42958262895358
// MI455X (gfx1250) — matrix operand order repaired and checked by reading, not compiled, not run on hardware
//
#include <hip/hip_runtime.h>

#define DIMN 2048
#define SEQ  2048
#define BATCH 2
#define NH   16
#define HD   128
#define MTOT (BATCH*SEQ)   // 4096

typedef __bf16 bf16_t;
typedef __bf16 v16bf __attribute__((ext_vector_type(16)));
typedef __bf16 v8bf  __attribute__((ext_vector_type(8)));
typedef float  v8f   __attribute__((ext_vector_type(8)));
typedef int    vi4   __attribute__((vector_size(16)));   // matches builtin proto

// ---------------------------------------------------------------------------
// Async global->LDS copy (gfx1250: GLOBAL_LOAD_ASYNC_TO_LDS_B128, ASYNCcnt),
// with a synchronous fallback if the toolchain lacks the builtins.
// ---------------------------------------------------------------------------
#if defined(__has_builtin)
#if __has_builtin(__builtin_amdgcn_global_load_async_to_lds_b128) && \
    __has_builtin(__builtin_amdgcn_s_wait_asynccnt)
#define HAVE_ASYNC_LDS 1
#endif
#endif
#ifndef HAVE_ASYNC_LDS
#define HAVE_ASYNC_LDS 0
#endif

// copy 16 bytes (one b128 lane transfer) from global to LDS
__device__ __forceinline__ void async_cp16(bf16_t* lds, const bf16_t* g) {
#if HAVE_ASYNC_LDS
  __builtin_amdgcn_global_load_async_to_lds_b128((vi4*)(void*)g, (vi4*)(void*)lds, 0, 0);
#else
  *(v8bf*)lds = *(const v8bf*)g;
#endif
}

template <int N>
__device__ __forceinline__ void async_wait() {
#if HAVE_ASYNC_LDS
  __builtin_amdgcn_s_wait_asynccnt((unsigned short)N);
#endif
}

__device__ __forceinline__ v8f wmma_bf16(v16bf a, v16bf b, v8f c) {
  // D = A(16x32 bf16) * B(32x16 bf16) + C(16x16 f32)
  return __builtin_amdgcn_wmma_f32_16x16x32_bf16(false, a, false, b, (short)0, c,
                                                 false, false);
}

__device__ __forceinline__ v8f zero8() {
  v8f z;
#pragma unroll
  for (int i = 0; i < 8; ++i) z[i] = 0.0f;
  return z;
}

// Assemble an A-matrix fragment: per-lane 16 bf16 = two adjacent 16B chunks
// (K 0..15 for lanes 0-15 and 16..31 for lanes 16-31; caller bakes the
// lane-half offset into p).
__device__ __forceinline__ v16bf load_a_frag(const bf16_t* p) {
  v8bf lo = *(const v8bf*)p;
  v8bf hi = *(const v8bf*)(p + 8);
  v16bf r;
#pragma unroll
  for (int i = 0; i < 8; ++i) { r[i] = lo[i]; r[8 + i] = hi[i]; }
  return r;
}

// ---------------------------------------------------------------------------
// fp32 -> bf16 bulk convert (x)
// ---------------------------------------------------------------------------
__global__ __launch_bounds__(256) void f32_to_bf16_kernel(const float* __restrict__ in,
                                                          bf16_t* __restrict__ out,
                                                          int n) {
  int i = (blockIdx.x * blockDim.x + threadIdx.x) * 4;
  if (i + 3 < n) {
    float4 v = *(const float4*)(in + i);
    out[i + 0] = (bf16_t)v.x;
    out[i + 1] = (bf16_t)v.y;
    out[i + 2] = (bf16_t)v.z;
    out[i + 3] = (bf16_t)v.w;
  }
}

// ---------------------------------------------------------------------------
// W [K=in][N=out] fp32  ->  Wt [N][K] bf16 (tiled transpose through LDS)
// ---------------------------------------------------------------------------
__global__ __launch_bounds__(256) void transpose_w_kernel(const float* __restrict__ W,
                                                          bf16_t* __restrict__ Wt) {
  __shared__ float tile[32][33];
  int x = blockIdx.x * 32 + threadIdx.x;   // column (out dim)
  int y0 = blockIdx.y * 32;                // row base (in dim)
#pragma unroll
  for (int j = threadIdx.y; j < 32; j += 8)
    tile[j][threadIdx.x] = W[(size_t)(y0 + j) * DIMN + x];
  __syncthreads();
  int xo = blockIdx.y * 32 + threadIdx.x;  // now K index
  int yo0 = blockIdx.x * 32;               // now N base
#pragma unroll
  for (int j = threadIdx.y; j < 32; j += 8)
    Wt[(size_t)(yo0 + j) * DIMN + xo] = (bf16_t)tile[threadIdx.x][j];
}

// ---------------------------------------------------------------------------
// WMMA GEMM: C[M=4096][N=2048] = A(bf16, row-major MxK) * Bt(bf16, NxK)^T + bias
// Block tile 128x128, K-step 32, double-buffered async LDS staging.
// 8 waves arranged 2(M)x4(N), each wave 64x32 = 4x2 WMMA tiles.
// mode 0: Q epilogue (bias + RoPE + 1/d scale, store [b,h,n,d] bf16)
// mode 1: K epilogue (bias + RoPE,            store [b,h,n,d] bf16)
// mode 2: V epilogue (bias,                   store [b,h,d,n] bf16, transposed)
// mode 3: out-proj   (bias,                   store f32 row-major to d_out)
// ---------------------------------------------------------------------------
struct GemmArgs {
  const bf16_t* A;
  const bf16_t* Bt0; const bf16_t* Bt1; const bf16_t* Bt2;
  const float*  b0;  const float*  b1;  const float*  b2;
  void* o0; void* o1; void* o2;
  int mode_base;
};

__global__ __launch_bounds__(256) void gemm_kernel(GemmArgs g) {
  const int z    = blockIdx.z;
  const int mode = g.mode_base + z;
  const bf16_t* Bt   = (z == 0) ? g.Bt0 : (z == 1) ? g.Bt1 : g.Bt2;
  const float*  bias = (z == 0) ? g.b0  : (z == 1) ? g.b1  : g.b2;
  void*         outp = (z == 0) ? g.o0  : (z == 1) ? g.o1  : g.o2;

  __shared__ __align__(64) bf16_t As[2][128 * 32];  // A tile, row stride 32
  __shared__ __align__(64) bf16_t Bs[2][128 * 32];  // Bt tile (n-major)

  const int tid  = threadIdx.x;
  const int lane = tid & 31;
  const int w    = tid >> 5;
  const int l16  = lane & 15;
  const int lh   = lane >> 4;          // lane half (0/1)
  const int wm   = w >> 2;             // 0..1 -> 64 rows each
  const int wn   = w & 3;              // 0..3 -> 32 cols each
  const int m0   = blockIdx.y * 128;
  const int n0   = blockIdx.x * 128;

  const int ldr_row = tid >> 1;        // 0..127
  const int ldr_kc  = (tid & 1) * 16;  // 0 or 16 (elements)

  const bf16_t* gA = g.A + (size_t)(m0 + ldr_row) * DIMN + ldr_kc;
  const bf16_t* gB = Bt + (size_t)(n0 + ldr_row) * DIMN + ldr_kc;

  // issue one K-tile (per thread: 32B of A + 32B of B = 4 async b128)
  auto issue_tile = [&](int kk, int buf) {
    bf16_t* la = &As[buf][ldr_row * 32 + ldr_kc];
    bf16_t* lb = &Bs[buf][ldr_row * 32 + ldr_kc];
    async_cp16(la,     gA + kk);
    async_cp16(la + 8, gA + kk + 8);
    async_cp16(lb,     gB + kk);
    async_cp16(lb + 8, gB + kk + 8);
  };

  v8f acc[4][2];
#pragma unroll
  for (int ms = 0; ms < 4; ++ms)
#pragma unroll
    for (int ns = 0; ns < 2; ++ns) acc[ms][ns] = zero8();

  issue_tile(0, 0);
  issue_tile(32, 1);

  for (int kk = 0; kk < DIMN; kk += 32) {
    const int buf = (kk >> 5) & 1;
    if (kk + 32 < DIMN) async_wait<4>(); else async_wait<0>();
    __syncthreads();  // this tile visible to all waves

    v16bf af[4], bfg[2];
#pragma unroll
    for (int ms = 0; ms < 4; ++ms)
      af[ms] = load_a_frag(&As[buf][(wm * 64 + ms * 16 + l16) * 32 + lh * 16]);
#pragma unroll
    for (int ns = 0; ns < 2; ++ns)
      bfg[ns] = *(const v16bf*)&Bs[buf][(wn * 32 + ns * 16 + l16) * 32 + lh * 16];
#pragma unroll
    for (int ms = 0; ms < 4; ++ms)
#pragma unroll
      for (int ns = 0; ns < 2; ++ns)
        acc[ms][ns] = wmma_bf16(af[ms], bfg[ns], acc[ms][ns]);

    __syncthreads();  // all waves done reading buf before it is overwritten
    if (kk + 64 < DIMN) issue_tile(kk + 64, buf);
  }

  // Epilogue. C/D layout: vgpr i -> row i (lanes 0-15) / row 8+i (lanes 16-31),
  // column = n-subtile base + lane%16.
#pragma unroll
  for (int ms = 0; ms < 4; ++ms) {
#pragma unroll
    for (int ns = 0; ns < 2; ++ns) {
      const int col = n0 + wn * 32 + ns * 16 + l16;
      const float bcol = bias[col];
#pragma unroll
      for (int i = 0; i < 8; ++i) {
        const int row = m0 + wm * 64 + ms * 16 + lh * 8 + i;
        float v = acc[ms][ns][i] + bcol;
        if (mode <= 1) {
          // interleaved RoPE: pairs (2j,2j+1) sit in adjacent lanes
          const int pos    = row & (SEQ - 1);
          const int dimIdx = col & (HD - 1);
          const float inv  = __powf(10000.0f, -(float)(dimIdx & ~1) * (1.0f / HD));
          const float ang  = (float)pos * inv;
          const float cs   = __cosf(ang);
          const float sn   = __sinf(ang);
          const float prt  = __shfl_xor(v, 1, 32);
          float rv = (col & 1) ? (v * cs + prt * sn) : (v * cs - prt * sn);
          if (mode == 0) rv *= (1.0f / (float)HD);  // q/sqrt(d) then logits/sqrt(d)
          const int b = row >> 11, h = col >> 7;
          ((bf16_t*)outp)[(size_t)(b * NH + h) * SEQ * HD + (size_t)pos * HD + dimIdx] =
              (bf16_t)rv;
        } else if (mode == 2) {
          const int pos    = row & (SEQ - 1);
          const int dimIdx = col & (HD - 1);
          const int b = row >> 11, h = col >> 7;
          ((bf16_t*)outp)[((size_t)(b * NH + h) * HD + dimIdx) * SEQ + pos] = (bf16_t)v;
        } else {
          ((float*)outp)[(size_t)row * DIMN + col] = v;
        }
      }
    }
  }
}

// ---------------------------------------------------------------------------
// Flash attention: Q [bh][n][d] bf16 (RoPE + 1/d pre-applied),
// K [bh][n][d] bf16 (RoPE), Vt [bh][d][n] bf16.  O [b*n][h*d] bf16.
// One wave = 16 query rows, full d=128 accumulator; 32-key steps.
// K/V tiles are shared by all 8 waves -> staged once per block in LDS via
// async copies, double-buffered.
// ---------------------------------------------------------------------------
__global__ __launch_bounds__(256) void attn_kernel(const bf16_t* __restrict__ Q,
                                                   const bf16_t* __restrict__ K,
                                                   const bf16_t* __restrict__ Vt,
                                                   bf16_t* __restrict__ O) {
  const int bh   = blockIdx.y;            // b*NH + h
  const int tid  = threadIdx.x;
  const int w    = tid >> 5;
  const int lane = tid & 31;
  const int l16  = lane & 15;
  const int lh   = lane >> 4;

  const bf16_t* Qh = Q + (size_t)bh * SEQ * HD;
  const bf16_t* Kh = K + (size_t)bh * SEQ * HD;
  const bf16_t* Vh = Vt + (size_t)bh * HD * SEQ;

  const int q0 = blockIdx.x * 128 + w * 16;

  // Q fragments: 4 chunks of K=32 over d=128, A layout (held in regs all loop)
  v16bf qf[4];
#pragma unroll
  for (int c = 0; c < 4; ++c)
    qf[c] = load_a_frag(Qh + (size_t)(q0 + l16) * HD + c * 32 + lh * 16);

  v8f acc[8];
#pragma unroll
  for (int nt = 0; nt < 8; ++nt) acc[nt] = zero8();
  float m[8], l[8];
#pragma unroll
  for (int i = 0; i < 8; ++i) { m[i] = -3.0e38f; l[i] = 0.0f; }

  __shared__ __align__(64) bf16_t Ks[2][32 * 128];   // 32 keys x 128 dims
  __shared__ __align__(64) bf16_t Vs[2][128 * 32];   // 128 dims x 32 keys
  __shared__ __align__(64) bf16_t pbuf[8][16 * 32];  // per-wave P tile

  // loader mapping: K tile 8KB -> 32B/thread; V tile 8KB -> 32B/thread
  const int krow = tid >> 3;          // 0..31  (key)
  const int kofs = (tid & 7) * 16;    // 0..112 (dim elements, 32B chunks)
  const int vrow = tid >> 1;          // 0..127 (dim)
  const int vofs = (tid & 1) * 16;    // 0/16   (key elements)

  auto issue_tile = [&](int k0, int buf) {
    const bf16_t* gk = Kh + (size_t)(k0 + krow) * HD + kofs;
    const bf16_t* gv = Vh + (size_t)vrow * SEQ + k0 + vofs;
    bf16_t* lk = &Ks[buf][krow * 128 + kofs];
    bf16_t* lv = &Vs[buf][vrow * 32 + vofs];
    async_cp16(lk,     gk);
    async_cp16(lk + 8, gk + 8);
    async_cp16(lv,     gv);
    async_cp16(lv + 8, gv + 8);
  };

  issue_tile(0, 0);
  issue_tile(32, 1);

  for (int k0 = 0; k0 < SEQ; k0 += 32) {
    const int buf = (k0 >> 5) & 1;
    if (k0 + 32 < SEQ) async_wait<4>(); else async_wait<0>();
    __syncthreads();

    // S(16x32) = Q(16x128) @ K^T(128x32): two 16-key tiles, 4 d-chunks each
    v8f s[2];
    s[0] = zero8();
    s[1] = zero8();
#pragma unroll
    for (int t = 0; t < 2; ++t)
#pragma unroll
      for (int c = 0; c < 4; ++c) {
        v16bf kf = *(const v16bf*)&Ks[buf][(t * 16 + l16) * 128 + c * 32 + lh * 16];
        s[t] = wmma_bf16(qf[c], kf, s[t]);
      }

    // online softmax: row = lh*8 + i lives in one VGPR across 16 lanes
    float f[8];
#pragma unroll
    for (int i = 0; i < 8; ++i) {
      float v = fmaxf(s[0][i], s[1][i]);
      v = fmaxf(v, __shfl_xor(v, 1, 32));
      v = fmaxf(v, __shfl_xor(v, 2, 32));
      v = fmaxf(v, __shfl_xor(v, 4, 32));
      v = fmaxf(v, __shfl_xor(v, 8, 32));
      const float mn = fmaxf(m[i], v);
      f[i] = __expf(m[i] - mn);
      m[i] = mn;
      const float p0 = __expf(s[0][i] - mn);
      const float p1 = __expf(s[1][i] - mn);
      s[0][i] = p0;
      s[1][i] = p1;
      float ps = p0 + p1;
      ps += __shfl_xor(ps, 1, 32);
      ps += __shfl_xor(ps, 2, 32);
      ps += __shfl_xor(ps, 4, 32);
      ps += __shfl_xor(ps, 8, 32);
      l[i] = l[i] * f[i] + ps;
    }
#pragma unroll
    for (int nt = 0; nt < 8; ++nt)
#pragma unroll
      for (int i = 0; i < 8; ++i) acc[nt][i] *= f[i];

    // Relayout P (C layout -> A layout) through this wave's private LDS slice;
    // per-wave DS ordering keeps store->load in order.
#pragma unroll
    for (int t = 0; t < 2; ++t)
#pragma unroll
      for (int i = 0; i < 8; ++i)
        pbuf[w][(lh * 8 + i) * 32 + t * 16 + l16] = (bf16_t)s[t][i];
    v16bf pf = load_a_frag(&pbuf[w][l16 * 32 + lh * 16]);

    // O += P(16x32) @ V(32x128): 8 n-tiles of 16 dims
#pragma unroll
    for (int nt = 0; nt < 8; ++nt) {
      v16bf vf = *(const v16bf*)&Vs[buf][(nt * 16 + l16) * 32 + lh * 16];
      acc[nt] = wmma_bf16(pf, vf, acc[nt]);
    }

    __syncthreads();  // everyone done reading Ks/Vs[buf] before overwrite
    if (k0 + 64 < SEQ) issue_tile(k0 + 64, buf);
  }

  const int b = bh >> 4, h = bh & 15;
#pragma unroll
  for (int nt = 0; nt < 8; ++nt) {
    const int dimIdx = nt * 16 + l16;
#pragma unroll
    for (int i = 0; i < 8; ++i) {
      const int pos = q0 + lh * 8 + i;
      O[((size_t)b * SEQ + pos) * DIMN + h * HD + dimIdx] = (bf16_t)(acc[nt][i] / l[i]);
    }
  }
}

// ---------------------------------------------------------------------------
// launch
// ---------------------------------------------------------------------------
extern "C" void kernel_launch(void* const* d_in, const int* in_sizes, int n_in,
                              void* d_out, int out_size, void* d_ws, size_t ws_size,
                              hipStream_t stream) {
  const float* x  = (const float*)d_in[0];
  const float* Wq = (const float*)d_in[1];
  const float* bq = (const float*)d_in[2];
  const float* Wk = (const float*)d_in[3];
  const float* bk = (const float*)d_in[4];
  const float* Wv = (const float*)d_in[5];
  const float* bv = (const float*)d_in[6];
  const float* Wo = (const float*)d_in[7];
  const float* bo = (const float*)d_in[8];

  char* ws = (char*)d_ws;
  const size_t MB = (size_t)1 << 20;
  bf16_t* xb  = (bf16_t*)(ws + 0 * MB);    // 16 MB  x in bf16
  bf16_t* Wqt = (bf16_t*)(ws + 16 * MB);   //  8 MB  each, transposed bf16
  bf16_t* Wkt = (bf16_t*)(ws + 24 * MB);
  bf16_t* Wvt = (bf16_t*)(ws + 32 * MB);
  bf16_t* Wot = (bf16_t*)(ws + 40 * MB);
  bf16_t* qb  = (bf16_t*)(ws + 48 * MB);   // 16 MB  [b,h,n,d]
  bf16_t* kb  = (bf16_t*)(ws + 64 * MB);   // 16 MB  [b,h,n,d]
  bf16_t* vb  = (bf16_t*)(ws + 80 * MB);   // 16 MB  [b,h,d,n]
  bf16_t* ob  = (bf16_t*)(ws + 96 * MB);   // 16 MB  attn output [b*n][h*d]

  // 1) convert x to bf16
  {
    const int n = MTOT * DIMN;
    f32_to_bf16_kernel<<<n / (256 * 4), 256, 0, stream>>>(x, xb, n);
  }
  // 2) transpose+convert the four weights
  {
    dim3 tg(DIMN / 32, DIMN / 32), tb(32, 8);
    transpose_w_kernel<<<tg, tb, 0, stream>>>(Wq, Wqt);
    transpose_w_kernel<<<tg, tb, 0, stream>>>(Wk, Wkt);
    transpose_w_kernel<<<tg, tb, 0, stream>>>(Wv, Wvt);
    transpose_w_kernel<<<tg, tb, 0, stream>>>(Wo, Wot);
  }
  // 3) fused QKV projection (+bias, +RoPE, +scale, V transposed)
  {
    GemmArgs a{xb, Wqt, Wkt, Wvt, bq, bk, bv, (void*)qb, (void*)kb, (void*)vb, 0};
    gemm_kernel<<<dim3(DIMN / 128, MTOT / 128, 3), 256, 0, stream>>>(a);
  }
  // 4) flash attention
  attn_kernel<<<dim3(SEQ / 128, BATCH * NH), 256, 0, stream>>>(qb, kb, vb, ob);
  // 5) output projection -> f32 d_out
  {
    GemmArgs a{ob, Wot, Wot, Wot, bo, bo, bo, d_out, d_out, d_out, 3};
    gemm_kernel<<<dim3(DIMN / 128, MTOT / 128, 1), 256, 0, stream>>>(a);
  }
}
